// MultiTowerGNN_9302899163664
// MI455X (gfx1250) — compile-verified, matrix operand order checked by reading, not run on hardware
//
#include <hip/hip_runtime.h>
#include <hip/hip_bf16.h>

// ---------------------------------------------------------------------------
// CDNA5 (gfx1250) WMMA types
// ---------------------------------------------------------------------------
typedef __attribute__((ext_vector_type(16))) _Float16 v16h;
typedef __attribute__((ext_vector_type(8)))  float    v8f;
typedef __attribute__((ext_vector_type(4)))  unsigned u32x4;

union HFrag { v16h v; u32x4 q[2]; };
union AccF  { v8f  v; float f[8]; };

// ---------------------------------------------------------------------------
// Ordered-uint mapping of float for atomicMax-based segment max
// ---------------------------------------------------------------------------
__device__ __forceinline__ unsigned ford(float f) {
  unsigned u = __float_as_uint(f);
  return (u & 0x80000000u) ? ~u : (u | 0x80000000u);
}
__device__ __forceinline__ float funord(unsigned o) {
  unsigned u = (o & 0x80000000u) ? (o ^ 0x80000000u) : ~o;
  return __uint_as_float(u);
}

// ---------------------------------------------------------------------------
// f32 -> f16 converts
// ---------------------------------------------------------------------------
__global__ void cvt_f16_kernel(const float* __restrict__ src, _Float16* __restrict__ dst,
                               size_t n) {
  size_t i = (size_t)blockIdx.x * blockDim.x + threadIdx.x;
  if (i < n) dst[i] = (_Float16)src[i];
}

// W [K x N] f32 row-major -> Wt [N x K] f16 (i.e. B^T, contiguous in K)
__global__ void cvtT_f16_kernel(const float* __restrict__ W, _Float16* __restrict__ Wt,
                                int K, int N) {
  size_t i = (size_t)blockIdx.x * blockDim.x + threadIdx.x;
  if (i >= (size_t)K * N) return;
  int k = (int)(i / N), n = (int)(i % N);
  Wt[(size_t)n * K + k] = (_Float16)W[i];
}

// src [M x C] f32 -> dst [M x LD] f16 at column offset (for concat build)
__global__ void cvt_cols_f16_kernel(const float* __restrict__ src, _Float16* __restrict__ dst,
                                    int M, int C, int LD, int off) {
  size_t i = (size_t)blockIdx.x * blockDim.x + threadIdx.x;
  if (i >= (size_t)M * C) return;
  int m = (int)(i / C), c = (int)(i % C);
  dst[(size_t)m * LD + off + c] = (_Float16)src[i];
}

// ---------------------------------------------------------------------------
// WMMA GEMM: C[MxN] = A[MxK] * B[KxN] (+bias, +add, activation)
// A: f16 row-major. Bt: f16 [N x K] (B transposed). C: f32 row-major.
// Requirements: K % 64 == 0, N % 64 == 0 (true for every GEMM here); M guarded.
// Each wave computes a 16x64 strip: one A fragment reused across 4 B fragments.
// The K-loop is software-pipelined with TWO register fragment sets so each
// WMMA group only waits for its own 10 b128 loads while the next set's 10
// loads are still in flight (partial s_wait_loadcnt instead of full drain).
// act: 0=none 1=relu 2=gelu(erf) 3=elu
// ---------------------------------------------------------------------------
struct FragSet { HFrag a, b0, b1, b2, b3; };

__device__ __forceinline__ void load_frags(FragSet& s,
                                           const _Float16* ap,  const _Float16* bp0,
                                           const _Float16* bp1, const _Float16* bp2,
                                           const _Float16* bp3, int koff) {
  s.a.q[0]  = *(const u32x4*)(ap  + koff);  s.a.q[1]  = *(const u32x4*)(ap  + koff + 16);
  s.b0.q[0] = *(const u32x4*)(bp0 + koff);  s.b0.q[1] = *(const u32x4*)(bp0 + koff + 16);
  s.b1.q[0] = *(const u32x4*)(bp1 + koff);  s.b1.q[1] = *(const u32x4*)(bp1 + koff + 16);
  s.b2.q[0] = *(const u32x4*)(bp2 + koff);  s.b2.q[1] = *(const u32x4*)(bp2 + koff + 16);
  s.b3.q[0] = *(const u32x4*)(bp3 + koff);  s.b3.q[1] = *(const u32x4*)(bp3 + koff + 16);
}

__device__ __forceinline__ void mma_frags(AccF* acc, const FragSet& s) {
  acc[0].v = __builtin_amdgcn_wmma_f32_16x16x32_f16(false, s.a.v, false, s.b0.v,
                                                    (short)0, acc[0].v, false, false);
  acc[1].v = __builtin_amdgcn_wmma_f32_16x16x32_f16(false, s.a.v, false, s.b1.v,
                                                    (short)0, acc[1].v, false, false);
  acc[2].v = __builtin_amdgcn_wmma_f32_16x16x32_f16(false, s.a.v, false, s.b2.v,
                                                    (short)0, acc[2].v, false, false);
  acc[3].v = __builtin_amdgcn_wmma_f32_16x16x32_f16(false, s.a.v, false, s.b3.v,
                                                    (short)0, acc[3].v, false, false);
}

__global__ __launch_bounds__(256)
void wmma_gemm_kernel(const _Float16* __restrict__ A,
                      const _Float16* __restrict__ Bt,
                      const float* __restrict__ bias,
                      const float* __restrict__ addsrc,
                      float* __restrict__ C,
                      int M, int N, int K, int act) {
  const int wave = threadIdx.x >> 5;
  const int lane = threadIdx.x & 31;
  const int rowBase = blockIdx.y * 128 + wave * 16;
  const int colBase = blockIdx.x * 64;

  int mA = rowBase + (lane & 15);
  if (mA >= M) mA = M - 1;             // branchless OOB (rows never stored)
  const int kb = (lane & 16) ? 8 : 0;  // lanes 16..31 hold the +8 / +24 K-halves

  const _Float16* ap  = A  + (size_t)mA * K + kb;
  const _Float16* bp0 = Bt + (size_t)(colBase +  0 + (lane & 15)) * K + kb;
  const _Float16* bp1 = Bt + (size_t)(colBase + 16 + (lane & 15)) * K + kb;
  const _Float16* bp2 = Bt + (size_t)(colBase + 32 + (lane & 15)) * K + kb;
  const _Float16* bp3 = Bt + (size_t)(colBase + 48 + (lane & 15)) * K + kb;

  AccF acc[4];
#pragma unroll
  for (int t = 0; t < 4; ++t)
#pragma unroll
    for (int r = 0; r < 8; ++r) acc[t].f[r] = 0.f;

  // Software-pipelined K loop, 2 register stages, 64 K per trip (K % 64 == 0).
  FragSet s0, s1;
  load_frags(s0, ap, bp0, bp1, bp2, bp3, 0);
  int k0 = 0;
  for (; k0 + 64 < K; k0 += 64) {
    load_frags(s1, ap, bp0, bp1, bp2, bp3, k0 + 32);  // prefetch stage 1
    mma_frags(acc, s0);                               // consume stage 0
    load_frags(s0, ap, bp0, bp1, bp2, bp3, k0 + 64);  // prefetch stage 0
    mma_frags(acc, s1);                               // consume stage 1
  }
  load_frags(s1, ap, bp0, bp1, bp2, bp3, K - 32);
  mma_frags(acc, s0);
  mma_frags(acc, s1);

  // C/D layout: VGPR r -> M=rowBase+r (lanes 0-15) or +8 (lanes 16-31); N=lane&15
  const int mrow = rowBase + ((lane & 16) ? 8 : 0);
  const int ncol = colBase + (lane & 15);
#pragma unroll
  for (int t = 0; t < 4; ++t) {
#pragma unroll
    for (int r = 0; r < 8; ++r) {
      const int m = mrow + r;
      if (m >= M) continue;
      const int n = ncol + t * 16;
      float v = acc[t].f[r];
      if (bias)   v += bias[n];
      if (addsrc) v += addsrc[(size_t)m * N + n];
      if (act == 1)      v = fmaxf(v, 0.f);
      else if (act == 2) v = 0.5f * v * (1.f + erff(v * 0.70710678118654752f));
      else if (act == 3) v = (v > 0.f) ? v : (expf(v) - 1.f);
      C[(size_t)m * N + n] = v;
    }
  }
}

// ---------------------------------------------------------------------------
// GATv2 edge pipeline (90k edges incl. self loops synthesized on the fly)
// ---------------------------------------------------------------------------
__global__ void edge_scores_kernel(const float* __restrict__ xl, const float* __restrict__ xr,
                                   const int* __restrict__ ei, const float* __restrict__ att,
                                   float* __restrict__ sc, unsigned* __restrict__ smax,
                                   int E0, int EE, int H, int C) {
  const int gw = (int)(((size_t)blockIdx.x * blockDim.x + threadIdx.x) >> 5);
  const int lane = threadIdx.x & 31;
  if (gw >= EE * H) return;
  const int e = gw / H, h = gw - e * H;
  const int s = (e < E0) ? ei[e] : (e - E0);
  const int d = (e < E0) ? ei[E0 + e] : (e - E0);
  const float* pl = xl + ((size_t)s * H + h) * C;
  const float* pr = xr + ((size_t)d * H + h) * C;
  const float* pa = att + (size_t)h * C;
  float sum = 0.f;
  for (int c = lane; c < C; c += 32) {
    float v = pl[c] + pr[c];
    v = (v > 0.f) ? v : 0.2f * v;   // leaky_relu(0.2)
    sum += v * pa[c];
  }
#pragma unroll
  for (int o = 16; o > 0; o >>= 1) sum += __shfl_down(sum, o, 32);
  if (lane == 0) {
    sc[(size_t)e * H + h] = sum;
    atomicMax(&smax[(size_t)d * H + h], ford(sum));
  }
}

__global__ void edge_exp_kernel(float* __restrict__ sc, const unsigned* __restrict__ smax,
                                float* __restrict__ den, const int* __restrict__ ei,
                                int E0, int EE, int H) {
  const int i = (int)((size_t)blockIdx.x * blockDim.x + threadIdx.x);
  if (i >= EE * H) return;
  const int e = i / H, h = i - e * H;
  const int d = (e < E0) ? ei[E0 + e] : (e - E0);
  const float ex = expf(sc[i] - funord(smax[(size_t)d * H + h]));
  sc[i] = ex;
  atomicAdd(&den[(size_t)d * H + h], ex);
}

__global__ void edge_scatter_kernel(const float* __restrict__ xl, const float* __restrict__ sc,
                                    const float* __restrict__ den, const int* __restrict__ ei,
                                    float* __restrict__ acc, int E0, int EE, int H, int C) {
  const int gw = (int)(((size_t)blockIdx.x * blockDim.x + threadIdx.x) >> 5);
  const int lane = threadIdx.x & 31;
  if (gw >= EE * H) return;
  const int e = gw / H, h = gw - e * H;
  const int s = (e < E0) ? ei[e] : (e - E0);
  const int d = (e < E0) ? ei[E0 + e] : (e - E0);
  const float alpha = sc[(size_t)e * H + h] / den[(size_t)d * H + h];
  const float* pl = xl + ((size_t)s * H + h) * C;
  float* pd = acc + ((size_t)d * H + h) * C;
  for (int c = lane; c < C; c += 32) atomicAdd(&pd[c], pl[c] * alpha);
}

// bias + ELU (+ eval BatchNorm) (+ residual)
__global__ void gat_epi_kernel(const float* __restrict__ acc, const float* __restrict__ bias,
                               const float* __restrict__ bng, const float* __restrict__ bnb,
                               const float* __restrict__ resid, float* __restrict__ out,
                               size_t total, int NC) {
  const size_t i = (size_t)blockIdx.x * blockDim.x + threadIdx.x;
  if (i >= total) return;
  const int n = (int)(i % NC);
  float v = acc[i] + bias[n];
  v = (v > 0.f) ? v : (expf(v) - 1.f);                       // ELU
  if (bng) v = bng[n] * (v * 0.9999950000374969f) + bnb[n];  // 1/sqrt(1+1e-5)
  if (resid) v += resid[i];
  out[i] = v;
}

// ---------------------------------------------------------------------------
// Host driver
// ---------------------------------------------------------------------------
extern "C" void kernel_launch(void* const* d_in, const int* in_sizes, int n_in,
                              void* d_out, int out_size, void* d_ws, size_t ws_size,
                              hipStream_t stream) {
  (void)in_sizes; (void)n_in; (void)out_size; (void)ws_size;
  const int M = 10000, E0 = 80000, EE = 90000;

  const float* x   = (const float*)d_in[0];
  const float* tag = (const float*)d_in[1];
  const float* txt = (const float*)d_in[2];
  const int*   ei  = (const int*)d_in[3];
  const float* g_wsrc[3] = {(const float*)d_in[4],  (const float*)d_in[8],  (const float*)d_in[12]};
  const float* g_wdst[3] = {(const float*)d_in[5],  (const float*)d_in[9],  (const float*)d_in[13]};
  const float* g_att[3]  = {(const float*)d_in[6],  (const float*)d_in[10], (const float*)d_in[14]};
  const float* g_bias[3] = {(const float*)d_in[7],  (const float*)d_in[11], (const float*)d_in[15]};
  const float* bn_g[2] = {(const float*)d_in[16], (const float*)d_in[18]};
  const float* bn_b[2] = {(const float*)d_in[17], (const float*)d_in[19]};
  const float* proj_w = (const float*)d_in[20]; const float* proj_b = (const float*)d_in[21];
  const float* tag_w1 = (const float*)d_in[22]; const float* tag_b1 = (const float*)d_in[23];
  const float* tag_w2 = (const float*)d_in[24]; const float* tag_b2 = (const float*)d_in[25];
  const float* txt_w1 = (const float*)d_in[26]; const float* txt_b1 = (const float*)d_in[27];
  const float* txt_w2 = (const float*)d_in[28]; const float* txt_b2 = (const float*)d_in[29];
  const float* fus_w1 = (const float*)d_in[30]; const float* fus_b1 = (const float*)d_in[31];
  const float* fus_w2 = (const float*)d_in[32]; const float* fus_b2 = (const float*)d_in[33];
  const float* pred_w = (const float*)d_in[34]; const float* pred_b = (const float*)d_in[35];

  float* out = (float*)d_out;
  float* z_struct = out;
  float* z_tag    = out + (size_t)M * 128;
  float* z_txt    = out + (size_t)M * 128 * 2;
  float* z_fus    = out + (size_t)M * 128 * 3;
  float* logits   = out + (size_t)M * 128 * 4;

  // ---- workspace carve (sequential stream -> safe reuse) ----
  char* p = (char*)d_ws;
  auto carve = [&](size_t bytes) -> void* {
    void* r = (void*)p; p += (bytes + 255) & ~(size_t)255; return r;
  };
  float*     xl   = (float*)carve((size_t)M * 2048 * 4);
  float*     xr   = (float*)carve((size_t)M * 2048 * 4);
  float*     acc  = (float*)carve((size_t)M * 2048 * 4);
  _Float16*  hb   = (_Float16*)carve((size_t)M * 2048 * 2);   // current activation, f16
  _Float16*  wt   = (_Float16*)carve((size_t)2048 * 2048 * 2);// transposed weight, f16
  _Float16*  tb   = (_Float16*)carve((size_t)M * 768 * 2);    // tower input, f16
  float*     th   = (float*)carve((size_t)M * 256 * 4);       // tower hidden, f32
  float*     sc   = (float*)carve((size_t)EE * 8 * 4);        // edge scores / exp
  unsigned*  smax = (unsigned*)carve((size_t)M * 8 * 4);
  float*     sden = (float*)carve((size_t)M * 8 * 4);

  auto cvt = [&](const float* srcf, _Float16* dstf, size_t n) {
    cvt_f16_kernel<<<(unsigned)((n + 255) / 256), 256, 0, stream>>>(srcf, dstf, n);
  };
  auto gemm = [&](const _Float16* Abf, const float* W, int K, int Ncols,
                  const float* bias, const float* add, float* Cout, int act) {
    size_t tot = (size_t)K * Ncols;
    cvtT_f16_kernel<<<(unsigned)((tot + 255) / 256), 256, 0, stream>>>(W, wt, K, Ncols);
    dim3 grid(Ncols / 64, (M + 127) / 128);
    wmma_gemm_kernel<<<grid, 256, 0, stream>>>(Abf, wt, bias, add, Cout, M, Ncols, K, act);
  };

  // =========================== GAT stack ===========================
  cvt(x, hb, (size_t)M * 256);
  for (int l = 0; l < 3; ++l) {
    const int K  = (l == 0) ? 256 : 2048;
    const int H  = (l < 2) ? 8 : 1;
    const int C  = (l < 2) ? 256 : 128;
    const int HC = H * C;

    gemm(hb, g_wsrc[l], K, HC, nullptr, nullptr, xl, 0);
    gemm(hb, g_wdst[l], K, HC, nullptr, nullptr, xr, 0);

    hipMemsetAsync(acc,  0, (size_t)M * HC * sizeof(float), stream);
    hipMemsetAsync(smax, 0, (size_t)M * H * sizeof(unsigned), stream);
    hipMemsetAsync(sden, 0, (size_t)M * H * sizeof(float), stream);

    const long long waves = (long long)EE * H;
    edge_scores_kernel<<<(unsigned)((waves * 32 + 255) / 256), 256, 0, stream>>>(
        xl, xr, ei, g_att[l], sc, smax, E0, EE, H, C);
    edge_exp_kernel<<<(unsigned)((waves + 255) / 256), 256, 0, stream>>>(
        sc, smax, sden, ei, E0, EE, H);
    edge_scatter_kernel<<<(unsigned)((waves * 32 + 255) / 256), 256, 0, stream>>>(
        xl, sc, sden, ei, acc, E0, EE, H, C);

    const float* resid = nullptr;
    if (l == 0) {  // residual projection of raw input (hb still holds f16(x))
      gemm(hb, proj_w, 256, 2048, proj_b, nullptr, xr, 0);  // xr free after scores
      resid = xr;
    }
    float* dstf = (l == 2) ? z_struct : xl;  // xl free after scatter
    gat_epi_kernel<<<(unsigned)(((size_t)M * HC + 255) / 256), 256, 0, stream>>>(
        acc, g_bias[l], (l < 2) ? bn_g[l] : nullptr, (l < 2) ? bn_b[l] : nullptr,
        resid, dstf, (size_t)M * HC, HC);
    if (l < 2) cvt(dstf, hb, (size_t)M * HC);
  }

  // =========================== towers ===========================
  // tag tower: 128 -> relu 256 -> 128
  cvt(tag, tb, (size_t)M * 128);
  gemm(tb, tag_w1, 128, 256, tag_b1, nullptr, th, 1);
  cvt(th, tb, (size_t)M * 256);
  gemm(tb, tag_w2, 256, 128, tag_b2, nullptr, z_tag, 0);

  // txt tower: 768 -> relu 128 -> 128
  cvt(txt, tb, (size_t)M * 768);
  gemm(tb, txt_w1, 768, 128, txt_b1, nullptr, th, 1);
  cvt(th, tb, (size_t)M * 128);
  gemm(tb, txt_w2, 128, 128, txt_b2, nullptr, z_txt, 0);

  // fusion: concat(z_struct, z_tag, z_txt) [Mx384] -> gelu 256 -> 128
  {
    const size_t n = (size_t)M * 128;
    cvt_cols_f16_kernel<<<(unsigned)((n + 255) / 256), 256, 0, stream>>>(z_struct, tb, M, 128, 384, 0);
    cvt_cols_f16_kernel<<<(unsigned)((n + 255) / 256), 256, 0, stream>>>(z_tag,    tb, M, 128, 384, 128);
    cvt_cols_f16_kernel<<<(unsigned)((n + 255) / 256), 256, 0, stream>>>(z_txt,    tb, M, 128, 384, 256);
  }
  gemm(tb, fus_w1, 384, 256, fus_b1, nullptr, th, 2);   // exact-erf GELU
  cvt(th, tb, (size_t)M * 256);
  gemm(tb, fus_w2, 256, 128, fus_b2, nullptr, z_fus, 0);

  // prediction head
  cvt(z_fus, tb, (size_t)M * 128);
  gemm(tb, pred_w, 128, 128, pred_b, nullptr, logits, 0);
}
